// DTNNStep_17085379904199
// MI455X (gfx1250) — matrix operand order repaired and checked by reading, not compiled, not run on hardware
//
#include <hip/hip_runtime.h>
#include <hip/hip_bf16.h>

typedef __bf16 bf16_t;
typedef __attribute__((ext_vector_type(16))) __bf16 v16bf;
typedef __attribute__((ext_vector_type(8)))  float  v8f;
typedef __attribute__((ext_vector_type(4)))  unsigned int u32x4;
typedef __attribute__((ext_vector_type(8)))  int i32x8;
typedef __attribute__((ext_vector_type(4)))  int i32x4;

#define N_ATOMS 50000
#define N_PAIRS 800000
#define N_EMB   128
#define N_DIST  100
#define N_HID   128

// workspace layout (bytes): three prepacked bf16 weight matrices + f32 afh
// NOTE: Wdf and Wfc are contiguous (64KB) so one TDM descriptor stages both.
#define WCF_OFF 0
#define WDF_OFF (32 * 1024)
#define WFC_OFF (64 * 1024)
#define AFH_OFF (96 * 1024)

#define PAIR_WAVES 4   // waves (pair tiles) per workgroup in pairs_kernel

__device__ __forceinline__ v8f wmma_bf16(v16bf a, v16bf b, v8f c) {
  // D = A(16x32 bf16) x B(32x16 bf16) + C(16x16 f32)
  return __builtin_amdgcn_wmma_f32_16x16x32_bf16(
      /*neg_a=*/false, a, /*neg_b=*/false, b,
      /*c_mod=*/(short)0, c, /*reuse_a=*/false, /*reuse_b=*/false);
}

// ---- Tensor Data Mover: 1-D copy global -> LDS (CDNA5 ISA) ----------
// nelem = number of 2-byte elements. Issued once per wave; caller must
// barrier before other waves read the LDS data.
__device__ __forceinline__ void tdm_load_1d(void* lds_dst, const void* gsrc,
                                            unsigned nelem) {
#if __has_builtin(__builtin_amdgcn_tensor_load_to_lds) && \
    __has_builtin(__builtin_amdgcn_s_wait_tensorcnt)
  unsigned long long ga = (unsigned long long)(uintptr_t)gsrc;
  unsigned lds_off = (unsigned)(uintptr_t)lds_dst;  // low 32b = LDS offset

  u32x4 g0;
  g0[0] = 1u;                                         // count=1 valid desc
  g0[1] = lds_off;                                    // lds_addr
  g0[2] = (unsigned)(ga & 0xFFFFFFFFu);               // global_addr[31:0]
  g0[3] = (unsigned)((ga >> 32) & 0x01FFFFFFu)        // global_addr[56:32]
          | 0x80000000u;                              // type=2 ("image")

  i32x8 g1;
  g1[0] = 0x00010000;                 // workgroup_mask=0, data_size=1 (2B)
  g1[1] = (int)((nelem & 0xFFFFu) << 16);       // tensor_dim0[15:0]
  g1[2] = (int)((nelem >> 16) & 0xFFFFu)        // tensor_dim0[31:16]
          | (1 << 16);                          // tensor_dim1 = 1
  g1[3] = (int)((nelem & 0xFFFFu) << 16);       // tile_dim0 = nelem
  g1[4] = 0x00010001;                           // tile_dim1=1, tile_dim2=1
  g1[5] = (int)nelem;                           // tensor_dim0_stride[31:0]
  g1[6] = (int)((nelem & 0xFFFFu) << 16);       // dim1_stride[15:0] (unused)
  g1[7] = (int)((nelem >> 16) & 0xFFFFu);       // dim1_stride[47:16]

  i32x4 g2;
  g2[0] = 1;                                    // tensor_dim2 = 1
  g2[1] = 1;                                    // tensor_dim3 = 1
  g2[2] = 0;                                    // tensor_dim2_stride lo
  g2[3] = 1 << 16;                              // tile_dim3 = 1

  i32x4 g3;
  g3[0] = 0;                                    // tensor_dim3_stride lo
  g3[1] = 1 << 16;                              // tensor_dim4[15:0] = 1
  g3[2] = 1 << 16;                              // tile_dim4 = 1
  g3[3] = 0;

  i32x8 g4 = {};                                // unused trailing group

  __builtin_amdgcn_tensor_load_to_lds(g0, g1, g2, g3, g4, /*cpol=*/0);
  __builtin_amdgcn_s_wait_tensorcnt(0);
#else
  // cooperative fallback: one wave copies nelem bf16 via 16B chunks
  const uint4* s = (const uint4*)gsrc;
  uint4* d = (uint4*)lds_dst;
  int lane = threadIdx.x & 31;
  for (unsigned i = lane; i < nelem / 8; i += 32) d[i] = s[i];
#endif
}

// ---- WMMA operand loaders (wave32 layouts, ISA 7.12.2) --------------------

// A-matrix 16x32 bf16 from row-major f32 [.. x ld], rows row..row+15,
// K slice [kbase, kbase+32), zero-pad K >= kmax.
__device__ __forceinline__ v16bf load_a_global(const float* __restrict__ src,
                                               int ld, int row, int kbase,
                                               int kmax) {
  int lane = threadIdx.x & 31;
  int r    = row + (lane & 15);
  int koff = (lane >> 4) * 8;
  const float* p = src + (long)r * ld;
  v16bf a;
#pragma unroll
  for (int v = 0; v < 8; ++v) {
    int k = kbase + koff + 2 * (v & 3) + ((v >> 2) * 16);
    float x0 = (k     < kmax) ? p[k]     : 0.0f;
    float x1 = (k + 1 < kmax) ? p[k + 1] : 0.0f;
    a[2 * v]     = (bf16_t)x0;
    a[2 * v + 1] = (bf16_t)x1;
  }
  return a;
}

// A-matrix 16x32 bf16 from a row-major bf16 LDS tile [16 x N_HID].
__device__ __forceinline__ v16bf load_a_lds(const bf16_t* lds, int kbase) {
  int lane = threadIdx.x & 31;
  int r    = lane & 15;
  int koff = (lane >> 4) * 8;
  v16bf a;
#pragma unroll
  for (int v = 0; v < 8; ++v) {
    int k = kbase + koff + 2 * (v & 3) + ((v >> 2) * 16);
    a[2 * v]     = lds[r * N_HID + k];
    a[2 * v + 1] = lds[r * N_HID + k + 1];
  }
  return a;
}

// B-matrix tile from a prepacked buffer (global or LDS): each lane's 16 bf16
// are contiguous (32B) -> 2x b128 loads.
__device__ __forceinline__ v16bf load_b(const bf16_t* packed, int tile) {
  int lane = threadIdx.x & 31;
  return *(const v16bf*)(packed + ((long)tile * 32 + lane) * 16);
}

// ---- kernel 1: prepack a weight matrix into B-layout bf16 -----------------
// W row-major [krows x 128] (zero-padded to 128 K rows). grid = 32 tiles
// (kt 0..3, nt 0..7), block = 32 lanes.
__global__ void __launch_bounds__(32) prepack_w(const float* __restrict__ W,
                                                int krows,
                                                bf16_t* __restrict__ dst) {
  int tt   = blockIdx.x;       // 0..31
  int kt   = tt >> 3;
  int nt   = tt & 7;
  int lane = threadIdx.x & 31;
  int n    = nt * 16 + (lane & 15);
  int kofs = (lane >> 4) * 8;
  v16bf b;
#pragma unroll
  for (int v = 0; v < 8; ++v) {
    int k = kt * 32 + kofs + 2 * (v & 3) + ((v >> 2) * 16);
    float x0 = (k     < krows) ? W[(long)k * N_HID + n]       : 0.0f;
    float x1 = (k + 1 < krows) ? W[(long)(k + 1) * N_HID + n] : 0.0f;
    b[2 * v]     = (bf16_t)x0;
    b[2 * v + 1] = (bf16_t)x1;
  }
  *(v16bf*)(dst + ((long)tt * 32 + lane) * 16) = b;
}

// ---- kernel 2: atom-side GEMMs -------------------------------------------
// afh = atom_features @ W_cf + b_cf               (stored f32 for gather)
// out = atom_features - tanh((b_df * afh) @ W_fc)
__global__ void __launch_bounds__(32) atoms_kernel(
    const float* __restrict__ atom_features, const bf16_t* __restrict__ Wcf_p,
    const bf16_t* __restrict__ Wfc_p, const float* __restrict__ b_cf,
    const float* __restrict__ b_df, float* __restrict__ afh,
    float* __restrict__ out) {
  __shared__ bf16_t lds_g[16 * N_HID];
  int lane  = threadIdx.x & 31;
  int m0    = blockIdx.x * 16;
  int ncol  = lane & 15;
  int mhalf = (lane >> 4) * 8;

  v8f acc[8];
#pragma unroll
  for (int nt = 0; nt < 8; ++nt) acc[nt] = (v8f){};

#pragma unroll
  for (int kt = 0; kt < 4; ++kt) {
    v16bf a = load_a_global(atom_features, N_EMB, m0, kt * 32, N_EMB);
#pragma unroll
    for (int nt = 0; nt < 8; ++nt)
      acc[nt] = wmma_bf16(a, load_b(Wcf_p, kt * 8 + nt), acc[nt]);
  }

  // bias, stash afh, stage g = b_df * afh into LDS (C-layout -> row major)
#pragma unroll
  for (int nt = 0; nt < 8; ++nt) {
    int n    = nt * 16 + ncol;
    float bc = b_cf[n];
    float bd = b_df[n];
#pragma unroll
    for (int v = 0; v < 8; ++v) {
      int m   = mhalf + v;
      float h = acc[nt][v] + bc;
      afh[(long)(m0 + m) * N_HID + n] = h;
      lds_g[m * N_HID + n]            = (bf16_t)(bd * h);
    }
  }
  __syncthreads();

  v8f acc2[8];
#pragma unroll
  for (int nt = 0; nt < 8; ++nt) acc2[nt] = (v8f){};
#pragma unroll
  for (int kt = 0; kt < 4; ++kt) {
    v16bf a = load_a_lds(lds_g, kt * 32);
#pragma unroll
    for (int nt = 0; nt < 8; ++nt)
      acc2[nt] = wmma_bf16(a, load_b(Wfc_p, kt * 8 + nt), acc2[nt]);
  }

#pragma unroll
  for (int nt = 0; nt < 8; ++nt) {
    int n = nt * 16 + ncol;
#pragma unroll
    for (int v = 0; v < 8; ++v) {
      long idx = (long)(m0 + mhalf + v) * N_EMB + n;
      out[idx] = atom_features[idx] - tanhf(acc2[nt][v]);
    }
  }
}

// ---- kernel 3: fused pair pipeline ---------------------------------------
// Per block: TDM-stage 64KB packed Wdf|Wfc into LDS once; 4 waves each own a
// 16-pair tile:
//   dh  = distance_tile @ W_df + b_df      (WMMA, B from LDS, K pad 100->128)
//   msg = dh * afh[j]                      (L2-resident gather)
//   t   = tanh(msg @ W_fc)                 (WMMA via LDS re-layout)
//   out[i] += t                            (f32 global atomics, i sorted)
__global__ void __launch_bounds__(32 * PAIR_WAVES) pairs_kernel(
    const float* __restrict__ distance, const int* __restrict__ dmi,
    const int* __restrict__ dmj, const bf16_t* __restrict__ Wdf_p,
    const float* __restrict__ b_df, const float* __restrict__ afh,
    float* __restrict__ out) {
  extern __shared__ char smem[];
  bf16_t* Wdf_l = (bf16_t*)smem;                       // 32KB
  bf16_t* Wfc_l = (bf16_t*)(smem + 32 * 1024);         // 32KB
  int wave = threadIdx.x >> 5;
  bf16_t* msg_l = (bf16_t*)(smem + 64 * 1024 + wave * 4096);  // 4KB/wave

  if (wave == 0) {
    // Wdf_p and Wfc_p are contiguous in the workspace: one 64KB TDM copy.
    tdm_load_1d(Wdf_l, Wdf_p, 32768u);
  }
  __syncthreads();

  int lane  = threadIdx.x & 31;
  int p0    = (blockIdx.x * PAIR_WAVES + wave) * 16;
  int ncol  = lane & 15;
  int mhalf = (lane >> 4) * 8;

  int jrow[8], irow[8];
#pragma unroll
  for (int v = 0; v < 8; ++v) {
    jrow[v] = dmj[p0 + mhalf + v];
    irow[v] = dmi[p0 + mhalf + v];
  }

  v8f acc[8];
#pragma unroll
  for (int nt = 0; nt < 8; ++nt) acc[nt] = (v8f){};
#pragma unroll
  for (int kt = 0; kt < 4; ++kt) {
    v16bf a = load_a_global(distance, N_DIST, p0, kt * 32, N_DIST);
#pragma unroll
    for (int nt = 0; nt < 8; ++nt)
      acc[nt] = wmma_bf16(a, load_b(Wdf_l, kt * 8 + nt), acc[nt]);
  }

#pragma unroll
  for (int nt = 0; nt < 8; ++nt) {
    int n    = nt * 16 + ncol;
    float bd = b_df[n];
#pragma unroll
    for (int v = 0; v < 8; ++v) {
      float msg = (acc[nt][v] + bd) * afh[(long)jrow[v] * N_HID + n];
      msg_l[(mhalf + v) * N_HID + n] = (bf16_t)msg;
    }
  }
  __syncthreads();  // uniform; also orders per-wave LDS write->read

  v8f acc2[8];
#pragma unroll
  for (int nt = 0; nt < 8; ++nt) acc2[nt] = (v8f){};
#pragma unroll
  for (int kt = 0; kt < 4; ++kt) {
    v16bf a = load_a_lds(msg_l, kt * 32);
#pragma unroll
    for (int nt = 0; nt < 8; ++nt)
      acc2[nt] = wmma_bf16(a, load_b(Wfc_l, kt * 8 + nt), acc2[nt]);
  }

#pragma unroll
  for (int nt = 0; nt < 8; ++nt) {
    int n = nt * 16 + ncol;
#pragma unroll
    for (int v = 0; v < 8; ++v) {
      unsafeAtomicAdd(&out[(long)irow[v] * N_EMB + n], tanhf(acc2[nt][v]));
    }
  }
}

extern "C" void kernel_launch(void* const* d_in, const int* in_sizes, int n_in,
                              void* d_out, int out_size, void* d_ws,
                              size_t ws_size, hipStream_t stream) {
  const float* atom_features = (const float*)d_in[0];
  const float* distance      = (const float*)d_in[1];
  // d_in[2] = atom_membership (unused by reference output)
  const int*   dmi = (const int*)d_in[3];
  const int*   dmj = (const int*)d_in[4];
  const float* Wcf = (const float*)d_in[5];
  const float* Wdf = (const float*)d_in[6];
  const float* Wfc = (const float*)d_in[7];
  const float* bcf = (const float*)d_in[8];
  const float* bdf = (const float*)d_in[9];
  float* out = (float*)d_out;

  char* ws      = (char*)d_ws;
  bf16_t* Wcf_p = (bf16_t*)(ws + WCF_OFF);
  bf16_t* Wdf_p = (bf16_t*)(ws + WDF_OFF);
  bf16_t* Wfc_p = (bf16_t*)(ws + WFC_OFF);
  float*  afh   = (float*)(ws + AFH_OFF);

  prepack_w<<<32, 32, 0, stream>>>(Wcf, N_EMB, Wcf_p);
  prepack_w<<<32, 32, 0, stream>>>(Wdf, N_DIST, Wdf_p);
  prepack_w<<<32, 32, 0, stream>>>(Wfc, N_HID, Wfc_p);

  atoms_kernel<<<N_ATOMS / 16, 32, 0, stream>>>(atom_features, Wcf_p, Wfc_p,
                                                bcf, bdf, afh, out);

  size_t smem_bytes = 64 * 1024 + PAIR_WAVES * 4096;  // 80KB / workgroup
  pairs_kernel<<<N_PAIRS / (16 * PAIR_WAVES), 32 * PAIR_WAVES, smem_bytes,
                 stream>>>(distance, dmi, dmj, Wdf_p, bdf, afh, out);
}
